// InvertedResidualBlockME_49469433316046
// MI455X (gfx1250) — matrix operand order repaired and checked by reading, not compiled, not run on hardware
//
#include <hip/hip_runtime.h>

#define CIN   32
#define H     192
#define COUT  32
#define KOFF  9
#define EPSV  1e-5f

typedef __bf16 bf16;
typedef __attribute__((ext_vector_type(16))) __bf16 v16bf;
typedef __attribute__((ext_vector_type(8)))  __bf16 v8bf;
typedef __attribute__((ext_vector_type(2)))  __bf16 v2bf;
typedef __attribute__((ext_vector_type(8)))  float  v8f;
typedef __attribute__((ext_vector_type(4)))  float  v4f;

union V16 { v16bf v; v8bf h[2]; };

// ---------------------------------------------------------------------------
// Kernel 1 (main): X1[row,192] = relu6(bn1(feats @ W1)) as bf16, full tiles
// only — no bounds checks anywhere. One wave = one 16-row tile; Cin=32 == one
// v_wmma_f32_16x16x32_bf16 K-step. W1 pre-swizzled in LDS into per-lane B
// fragments with the bn1 scale folded in.
// ---------------------------------------------------------------------------
__global__ __launch_bounds__(256)
void expand_kernel(const float* __restrict__ feats, const float* __restrict__ W1,
                   const float* __restrict__ g1, const float* __restrict__ b1,
                   const float* __restrict__ m1, const float* __restrict__ v1,
                   bf16* __restrict__ X1, int nfull)
{
    __shared__ __align__(16) bf16 sW1B[12][32][16]; // 12 KB, swizzled B frags
    __shared__ float sScale[H];
    __shared__ float sShift[H];

    const int t = threadIdx.x;
    if (t < H) {
        float sc  = g1[t] * rsqrtf(v1[t] + EPSV);
        sScale[t] = sc;
        sShift[t] = b1[t] - m1[t] * sc;
    }
    __syncthreads();

    // B fragment layout: lane l (col = ct*16 + (l&15)), element e -> K = 8*(l>>4)+(e&7)+16*(e>>3)
    for (int i = t; i < 12 * 32 * 16; i += 256) {
        int ct   = i >> 9;
        int lane = (i >> 4) & 31;
        int e    = i & 15;
        int col  = ct * 16 + (lane & 15);
        int krow = 8 * (lane >> 4) + (e & 7) + ((e >> 3) << 4);
        sW1B[ct][lane][e] = (bf16)(W1[krow * H + col] * sScale[col]);
    }
    __syncthreads();

    // Scalar (SGPR) wave id so per-wave control flow is s_cbranch, not exec ops.
    const int wave = __builtin_amdgcn_readfirstlane(t) >> 5;
    const int lane = t & 31;
    const int l16  = lane & 15;
    const int g    = lane >> 4;

    const int tile = blockIdx.x * 8 + wave;          // scalar
    if (tile >= nfull) return;                       // scalar branch

    const int row = tile * 16 + l16;
    const float* fr = feats + (long)row * CIN;

    // A fragment (16-bit A 16x32 layout): elems 0..7 -> K = 8g..8g+7,
    //                                     elems 8..15 -> K = 16+8g..16+8g+7
    V16 A;
    v4f lo0 = *(const v4f*)(fr + 8 * g);
    v4f lo1 = *(const v4f*)(fr + 8 * g + 4);
    v4f hi0 = *(const v4f*)(fr + 16 + 8 * g);
    v4f hi1 = *(const v4f*)(fr + 16 + 8 * g + 4);
#pragma unroll
    for (int e = 0; e < 4; ++e) {
        A.v[e]      = (bf16)lo0[e];
        A.v[e + 4]  = (bf16)lo1[e];
        A.v[e + 8]  = (bf16)hi0[e];
        A.v[e + 12] = (bf16)hi1[e];
    }

#pragma unroll
    for (int ct = 0; ct < 12; ++ct) {
        const int col = ct * 16 + l16;
        V16 B;
        B.h[0] = *(const v8bf*)&sW1B[ct][lane][0];
        B.h[1] = *(const v8bf*)&sW1B[ct][lane][8];

        v8f acc = {};
        acc = __builtin_amdgcn_wmma_f32_16x16x32_bf16(
                  false, A.v, false, B.v, (short)0, acc, false, false);

        const float sh = sShift[col];
        bf16* op = X1 + (long)(tile * 16 + 8 * g) * H + col;
#pragma unroll
        for (int r = 0; r < 8; ++r) {
            float y = fminf(fmaxf(acc[r] + sh, 0.0f), 6.0f);
            op[(long)r * H] = (bf16)y;
        }
    }
}

// ---------------------------------------------------------------------------
// Kernel 2 (main): depthwise sparse gather (+bn2/relu6, scale folded into W2)
// into LDS bf16, then per-wave 16x32 GEMM vs scale3-folded W3 (6 chained
// WMMAs) + shift3 + residual. Full tiles only. Block = 256 threads, 64 rows.
// ---------------------------------------------------------------------------
__global__ __launch_bounds__(256)
void dw_project_kernel(const float* __restrict__ feats,
                       const int*   __restrict__ nbr,
                       const float* __restrict__ W2,
                       const float* __restrict__ W3,
                       const float* __restrict__ g2, const float* __restrict__ b2,
                       const float* __restrict__ m2, const float* __restrict__ v2,
                       const float* __restrict__ g3, const float* __restrict__ b3,
                       const float* __restrict__ m3, const float* __restrict__ v3,
                       const bf16*  __restrict__ X1,
                       float* __restrict__ out, int N, int nfull)
{
    __shared__ __align__(16) bf16 sY[64][H];          // 24 KB
    __shared__ __align__(16) bf16 sW3B[2][6][32][16]; // 12 KB, swizzled B frags
    __shared__ float sW2[KOFF][H];                    // 6.75 KB (scale2 folded)
    __shared__ int   sNbr[64][KOFF];                  // 2.25 KB
    __shared__ float sShift2[H];
    __shared__ float sScale2[H];
    __shared__ float sScale3[COUT], sShift3[COUT];

    const int t    = threadIdx.x;
    const int row0 = blockIdx.x * 64;

    if (t < H) {
        float sc   = g2[t] * rsqrtf(v2[t] + EPSV);
        sScale2[t] = sc;
        sShift2[t] = b2[t] - m2[t] * sc;
    }
    if (t < COUT) {
        float sc   = g3[t] * rsqrtf(v3[t] + EPSV);
        sScale3[t] = sc;
        sShift3[t] = b3[t] - m3[t] * sc;
    }
    __syncthreads();

    for (int i = t; i < 2 * 6 * 32 * 16; i += 256) {
        int ct   = i / 3072;
        int rem  = i - ct * 3072;
        int kk   = rem >> 9;
        int lane = (rem >> 4) & 31;
        int e    = i & 15;
        int col  = ct * 16 + (lane & 15);
        int krow = kk * 32 + 8 * (lane >> 4) + (e & 7) + ((e >> 3) << 4);
        sW3B[ct][kk][lane][e] = (bf16)(W3[krow * COUT + col] * sScale3[col]);
    }
    for (int i = t; i < KOFF * H; i += 256) {
        int c = i % H;
        sW2[i / H][c] = W2[i] * sScale2[c];
    }
    for (int i = t; i < 64 * KOFF; i += 256) {
        int r = i / KOFF, k = i - r * KOFF;
        int grow   = row0 + r;
        sNbr[r][k] = (grow < N) ? nbr[(long)grow * KOFF + k] : -1;
    }
    __syncthreads();

    // Scalar (SGPR) wave id so per-wave control flow is s_cbranch, not exec ops.
    const int wave = __builtin_amdgcn_readfirstlane(t) >> 5;
    const int lane = t & 31;

    // ---- Phase 1: depthwise gather. Wave owns 8 rows; lane covers channel
    // pairs c0, c0+64, c0+128. Neighbor index is row-uniform -> SGPR base,
    // gathers become saddr + voffset b32 loads from L2-resident X1.
    {
        const int c0 = 2 * lane;
        float wreg[KOFF][6];
#pragma unroll
        for (int k = 0; k < KOFF; ++k) {
            const float* wp = &sW2[k][c0];
            wreg[k][0] = wp[0];   wreg[k][1] = wp[1];
            wreg[k][2] = wp[64];  wreg[k][3] = wp[65];
            wreg[k][4] = wp[128]; wreg[k][5] = wp[129];
        }
        for (int rr = 0; rr < 8; ++rr) {
            const int r = wave * 8 + rr;             // scalar
            float a0 = 0.f, a1 = 0.f, a2 = 0.f, a3 = 0.f, a4 = 0.f, a5 = 0.f;
#pragma unroll
            for (int k = 0; k < KOFF; ++k) {
                const int nb = __builtin_amdgcn_readfirstlane(sNbr[r][k]);
                if (nb >= 0) {                       // scalar branch
                    const bf16* xr = X1 + (long)nb * H;
                    v2bf x0 = *(const v2bf*)(xr + c0);
                    v2bf x1 = *(const v2bf*)(xr + c0 + 64);
                    v2bf x2 = *(const v2bf*)(xr + c0 + 128);
                    a0 += (float)x0[0] * wreg[k][0];
                    a1 += (float)x0[1] * wreg[k][1];
                    a2 += (float)x1[0] * wreg[k][2];
                    a3 += (float)x1[1] * wreg[k][3];
                    a4 += (float)x2[0] * wreg[k][4];
                    a5 += (float)x2[1] * wreg[k][5];
                }
            }
            float acc[6] = { a0, a1, a2, a3, a4, a5 };
#pragma unroll
            for (int ch = 0; ch < 3; ++ch) {
                const int c = c0 + 64 * ch;
                float y0 = fminf(fmaxf(acc[2 * ch]     + sShift2[c],     0.f), 6.f);
                float y1 = fminf(fmaxf(acc[2 * ch + 1] + sShift2[c + 1], 0.f), 6.f);
                v2bf o;
                o[0] = (bf16)y0;
                o[1] = (bf16)y1;
                *(v2bf*)&sY[r][c] = o;
            }
        }
    }
    __syncthreads();

    // ---- Phase 2: per-wave 16x192 @ 192x32 GEMM, 6 chained WMMAs per col tile.
    const int l16  = lane & 15;
    const int g    = lane >> 4;
    const int tile = blockIdx.x * 4 + wave;          // scalar
    if (wave >= 4 || tile >= nfull) return;          // scalar branch

    const int rloc = wave * 16 + l16;
    V16 A[6];
#pragma unroll
    for (int kk = 0; kk < 6; ++kk) {
        const bf16* p = &sY[rloc][kk * 32 + 8 * g];
        A[kk].h[0] = *(const v8bf*)p;
        A[kk].h[1] = *(const v8bf*)(p + 16);
    }

#pragma unroll
    for (int ct = 0; ct < 2; ++ct) {
        const int col = ct * 16 + l16;
        v8f acc = {};
#pragma unroll
        for (int kk = 0; kk < 6; ++kk) {
            V16 B;
            B.h[0] = *(const v8bf*)&sW3B[ct][kk][lane][0];
            B.h[1] = *(const v8bf*)&sW3B[ct][kk][lane][8];
            acc = __builtin_amdgcn_wmma_f32_16x16x32_bf16(
                      false, A[kk].v, false, B.v, (short)0, acc, false, false);
        }
        const float sh = sShift3[col];
        const int rbase = tile * 16 + 8 * g;
        float*       op = out   + (long)rbase * COUT + col;
        const float* fp = feats + (long)rbase * COUT + col;
#pragma unroll
        for (int r = 0; r < 8; ++r)
            op[(long)r * COUT] = acc[r] + sh + fp[(long)r * COUT];
    }
}

// ---------------------------------------------------------------------------
// Tail kernels: handle the <16 remainder rows with plain scalar code so the
// hot kernels carry zero bounds checks. Launched only when N % 16 != 0.
// ---------------------------------------------------------------------------
__global__ void tail_expand_kernel(const float* __restrict__ feats,
                                   const float* __restrict__ W1,
                                   const float* __restrict__ g1, const float* __restrict__ b1,
                                   const float* __restrict__ m1, const float* __restrict__ v1,
                                   bf16* __restrict__ X1, int N, int rstart)
{
    int idx = blockIdx.x * blockDim.x + threadIdx.x;     // (row-rstart)*H + c
    int total = (N - rstart) * H;
    if (idx >= total) return;
    int row = rstart + idx / H;
    int c   = idx % H;
    float acc = 0.f;
    for (int k = 0; k < CIN; ++k)
        acc += feats[(long)row * CIN + k] * W1[k * H + c];
    float sc = g1[c] * rsqrtf(v1[c] + EPSV);
    float y  = acc * sc + (b1[c] - m1[c] * sc);
    X1[(long)row * H + c] = (bf16)fminf(fmaxf(y, 0.f), 6.f);
}

__global__ void tail_project_kernel(const float* __restrict__ feats,
                                    const int*   __restrict__ nbr,
                                    const float* __restrict__ W2,
                                    const float* __restrict__ W3,
                                    const float* __restrict__ g2, const float* __restrict__ b2,
                                    const float* __restrict__ m2, const float* __restrict__ v2,
                                    const float* __restrict__ g3, const float* __restrict__ b3,
                                    const float* __restrict__ m3, const float* __restrict__ v3,
                                    const bf16* __restrict__ X1,
                                    float* __restrict__ out, int N, int rstart)
{
    int idx = blockIdx.x * blockDim.x + threadIdx.x;     // (row-rstart)*COUT + co
    int total = (N - rstart) * COUT;
    if (idx >= total) return;
    int row = rstart + idx / COUT;
    int co  = idx % COUT;
    float acc = 0.f;
    for (int c = 0; c < H; ++c) {
        float a = 0.f;
        for (int k = 0; k < KOFF; ++k) {
            int nb = nbr[(long)row * KOFF + k];
            if (nb >= 0) a += (float)X1[(long)nb * H + c] * W2[k * H + c];
        }
        float sc2 = g2[c] * rsqrtf(v2[c] + EPSV);
        a = a * sc2 + (b2[c] - m2[c] * sc2);
        a = fminf(fmaxf(a, 0.f), 6.f);
        acc += a * W3[c * COUT + co];
    }
    float sc3 = g3[co] * rsqrtf(v3[co] + EPSV);
    out[(long)row * COUT + co] =
        acc * sc3 + (b3[co] - m3[co] * sc3) + feats[(long)row * COUT + co];
}

// ---------------------------------------------------------------------------
extern "C" void kernel_launch(void* const* d_in, const int* in_sizes, int n_in,
                              void* d_out, int out_size, void* d_ws, size_t ws_size,
                              hipStream_t stream)
{
    const float* feats = (const float*)d_in[0];
    const int*   nbr   = (const int*)  d_in[1];
    const float* W1    = (const float*)d_in[2];
    const float* W2    = (const float*)d_in[3];
    const float* W3    = (const float*)d_in[4];
    const float* g1 = (const float*)d_in[5];
    const float* b1 = (const float*)d_in[6];
    const float* m1 = (const float*)d_in[7];
    const float* v1 = (const float*)d_in[8];
    const float* g2 = (const float*)d_in[9];
    const float* b2 = (const float*)d_in[10];
    const float* m2 = (const float*)d_in[11];
    const float* v2 = (const float*)d_in[12];
    const float* g3 = (const float*)d_in[13];
    const float* b3 = (const float*)d_in[14];
    const float* m3 = (const float*)d_in[15];
    const float* v3 = (const float*)d_in[16];

    float* out = (float*)d_out;
    const int N      = in_sizes[0] / CIN;
    const int nfull  = N / 16;             // full 16-row tiles
    const int rstart = nfull * 16;         // first remainder row
    const int ntail  = N - rstart;

    bf16* X1 = (bf16*)d_ws;   // N * 192 * 2 bytes = 76.8 MB for N=200000

    if (nfull > 0) {
        const int blocks1 = (nfull + 7) / 8;
        expand_kernel<<<blocks1, 256, 0, stream>>>(feats, W1, g1, b1, m1, v1,
                                                   X1, nfull);
    }
    if (ntail > 0) {
        const int tb = (ntail * H + 255) / 256;
        tail_expand_kernel<<<tb, 256, 0, stream>>>(feats, W1, g1, b1, m1, v1,
                                                   X1, N, rstart);
    }
    if (nfull > 0) {
        const int blocks2 = (nfull + 3) / 4;
        dw_project_kernel<<<blocks2, 256, 0, stream>>>(feats, nbr, W2, W3,
                                                       g2, b2, m2, v2,
                                                       g3, b3, m3, v3,
                                                       X1, out, N, nfull);
    }
    if (ntail > 0) {
        const int tb = (ntail * COUT + 255) / 256;
        tail_project_kernel<<<tb, 256, 0, stream>>>(feats, nbr, W2, W3,
                                                    g2, b2, m2, v2,
                                                    g3, b3, m3, v3,
                                                    X1, out, N, rstart);
    }
}
